// MeshGraphNet_30262339567815
// MI455X (gfx1250) — matrix operand order repaired and checked by reading, not compiled, not run on hardware
//
#include <hip/hip_runtime.h>

// ---------------------------------------------------------------------------
// MeshGraphNet on gfx1250: bf16 WMMA (v_wmma_f32_16x16x32_bf16) MLP blocks,
// fused gather / LayerNorm / residual / scatter-add epilogues.
// Specialized template variants keep the A-fragment path branch-free;
// fragment packing uses packed f32->bf16 conversion (vector fptrunc).
// ---------------------------------------------------------------------------

typedef __attribute__((ext_vector_type(16))) __bf16    bf16x16;
typedef __attribute__((ext_vector_type(2)))  __bf16    bf16x2;
typedef __attribute__((ext_vector_type(8)))  float     f32x8;
typedef __attribute__((ext_vector_type(4)))  float     f32x4;
typedef __attribute__((ext_vector_type(2)))  float     f32x2;
typedef __attribute__((ext_vector_type(8)))  unsigned  u32x8;

#define LATENT            128
#define LSTR              132   // LDS row stride (floats); keeps lane halves on different banks
#define WAVES_PER_BLOCK   4
#define TB                (WAVES_PER_BLOCK * 32)
#define NJOBS             39

// float -> bf16 (RNE bit trick; used by the weight-prep kernel)
__device__ __forceinline__ unsigned bf_bits(float f) {
    unsigned u = __builtin_bit_cast(unsigned, f);
    u += 0x7fffu + ((u >> 16) & 1u);
    return u >> 16;
}
__device__ __forceinline__ __bf16 f2bf(float f) {
    unsigned short hs = (unsigned short)bf_bits(f);
    return __builtin_bit_cast(__bf16, hs);
}
// pack two floats into one dword of two bf16
__device__ __forceinline__ unsigned pkbf(float f0, float f1) {
#if __has_builtin(__builtin_amdgcn_cvt_pk_bf16_f32)
    return __builtin_bit_cast(unsigned, __builtin_amdgcn_cvt_pk_bf16_f32(f0, f1));
#else
    f32x2 f; f[0] = f0; f[1] = f1;
    bf16x2 b = __builtin_convertvector(f, bf16x2);   // vector fptrunc -> packed cvt if HW has it
    return __builtin_bit_cast(unsigned, b);
#endif
}

// ---------------------------------------------------------------------------
// Weight prep: fp32 [K][N] -> bf16 [Kpad][Npad], zero padded.
// ---------------------------------------------------------------------------
struct WJob { const float* src; __bf16* dst; int K, N, Kpad, Npad; };
struct PrepArgs { WJob jobs[NJOBS]; };

__global__ void prep_weights_kernel(PrepArgs P) {
    WJob j = P.jobs[blockIdx.x];
    int tot = j.Kpad * j.Npad;
    int stride = gridDim.y * blockDim.x;
    for (int i = blockIdx.y * blockDim.x + threadIdx.x; i < tot; i += stride) {
        int k = i / j.Npad, n = i - k * j.Npad;
        float v = (k < j.K && n < j.N) ? j.src[k * j.N + n] : 0.f;
        j.dst[i] = f2bf(v);
    }
}

__global__ void fill_zero_kernel(float* p, long n) {
    long i = (long)blockIdx.x * blockDim.x + threadIdx.x;
    long stride = (long)gridDim.x * blockDim.x;
    for (; i < n; i += stride) p[i] = 0.f;
}

// ---------------------------------------------------------------------------
// Fused 3-layer MLP block (K -> 128 -> 128 -> N3), one wave = one 16-row tile.
// AVAR: 0 = single narrow segment (encoders, K padded to 32)
//       1 = three 128-wide segments with optional gather (edge / node blocks)
//       2 = single 128-wide identity segment (decoder)
// MODE: 0 out=y    1 out+=y    2 out+=y and atomicAdd(agg[scat[row]], y)
// ---------------------------------------------------------------------------
struct Seg { const float* ptr; const int* idx; int kstart; int width; int ld; };

struct BlockArgs {
    Seg seg[3]; int rows;
    const __bf16* W1; const __bf16* W2; const __bf16* W3;
    const float* b1; const float* b2; const float* b3;
    const float* ln_g; const float* ln_b;
    int N3;
    float* out; int out_ld;
    float* agg; const int* scat;
};

template<int AVAR, int KPAD, int NT3, bool LN, int MODE>
__global__ __launch_bounds__(TB) void mlp_block_kernel(BlockArgs A) {
    __shared__ float sm_all[WAVES_PER_BLOCK][16 * LSTR];
    const int wv   = threadIdx.x >> 5;
    const int lane = threadIdx.x & 31;
    const int m    = lane & 15;
    const int h    = lane >> 4;
    const int tile0 = (blockIdx.x * WAVES_PER_BLOCK + wv) * 16;
    float* sm = sm_all[wv];

    int grow = tile0 + m;
    int gr = grow < A.rows ? grow : (A.rows - 1);

    // unconditional row pointers (no nullptr phi -> global addrspace inference)
    const float* rp0;
    const float* rp1 = nullptr;
    const float* rp2 = nullptr;
    if constexpr (AVAR == 1) {
        int r0 = A.seg[0].idx ? A.seg[0].idx[gr] : gr;
        int r1 = A.seg[1].idx ? A.seg[1].idx[gr] : gr;
        rp0 = A.seg[0].ptr + (size_t)r0 * LATENT;
        rp1 = A.seg[1].ptr + (size_t)r1 * LATENT;
        rp2 = A.seg[2].ptr + (size_t)gr * LATENT;
    } else if constexpr (AVAR == 2) {
        rp0 = A.seg[0].ptr + (size_t)gr * LATENT;
    } else {
        rp0 = A.seg[0].ptr + (size_t)gr * A.seg[0].ld;
    }

    // ---- A-fragment builders (ISA 16-bit A layout: k = 8h + (j<8 ? j : 8+j)) --
    auto build_a_l1 = [&](int kb) -> bf16x16 {
        u32x8 au;
        if constexpr (AVAR == 0) {
            const int w = A.seg[0].width;
            float v[16];
#pragma unroll
            for (int j = 0; j < 16; ++j) {
                int k = kb + 8 * h + (j < 8 ? j : 8 + j);
                int kc = k < w ? k : (w - 1);        // clamped, always-valid load
                float t = rp0[kc];
                v[j] = k < w ? t : 0.f;
            }
#pragma unroll
            for (int j = 0; j < 8; ++j) au[j] = pkbf(v[2 * j], v[2 * j + 1]);
        } else {
            const float* base;
            if constexpr (AVAR == 1) {
                int s = kb >> 7;                     // chunk never crosses a segment
                const float* rs_ = (s == 0) ? rp0 : (s == 1 ? rp1 : rp2);
                base = rs_ + (kb & 127) + 8 * h;
            } else {
                base = rp0 + kb + 8 * h;
            }
            f32x8 r0 = *(const f32x8*)base;          // k = kb+8h   .. +7
            f32x8 r1 = *(const f32x8*)(base + 16);   // k = kb+8h+16 .. +23
#pragma unroll
            for (int j = 0; j < 4; ++j) {
                au[j]     = pkbf(r0[2 * j], r0[2 * j + 1]);
                au[4 + j] = pkbf(r1[2 * j], r1[2 * j + 1]);
            }
        }
        return __builtin_bit_cast(bf16x16, au);
    };
    auto build_a_lds = [&](int kb) -> bf16x16 {
        const float* p = &sm[m * LSTR + kb + 8 * h];
        f32x4 q0 = *(const f32x4*)p;
        f32x4 q1 = *(const f32x4*)(p + 4);
        f32x4 q2 = *(const f32x4*)(p + 16);
        f32x4 q3 = *(const f32x4*)(p + 20);
        u32x8 au;
        au[0] = pkbf(q0[0], q0[1]); au[1] = pkbf(q0[2], q0[3]);
        au[2] = pkbf(q1[0], q1[1]); au[3] = pkbf(q1[2], q1[3]);
        au[4] = pkbf(q2[0], q2[1]); au[5] = pkbf(q2[2], q2[3]);
        au[6] = pkbf(q3[0], q3[1]); au[7] = pkbf(q3[2], q3[3]);
        return __builtin_bit_cast(bf16x16, au);
    };

    f32x8 acc[8];
    auto zero_acc = [&]() {
#pragma unroll
        for (int nt = 0; nt < 8; ++nt)
#pragma unroll
            for (int t = 0; t < 8; ++t) acc[nt][t] = 0.f;
    };
    // bias(+relu) -> LDS; C layout: VGPR v holds row v+8h, col nt*16+m
    auto acc_to_lds = [&](const float* bias, bool relu) {
#pragma unroll
        for (int nt = 0; nt < 8; ++nt) {
            int col = nt * 16 + m;
            float bb = bias[col];
#pragma unroll
            for (int v = 0; v < 8; ++v) {
                float x = acc[nt][v] + bb;
                sm[(v + 8 * h) * LSTR + col] = (relu && x < 0.f) ? 0.f : x;
            }
        }
    };

    // ---------------- layer 1: [16 x KPAD] @ [KPAD x 128] ----------------
    zero_acc();
    for (int kb = 0; kb < KPAD; kb += 32) {
        bf16x16 a = build_a_l1(kb);
#pragma unroll
        for (int nt = 0; nt < 8; ++nt) {
            bf16x16 b;
#pragma unroll
            for (int s = 0; s < 16; ++s)
                b[s] = A.W1[(size_t)(kb + 8 * h + (s < 8 ? s : 8 + s)) * 128 + nt * 16 + m];
            acc[nt] = __builtin_amdgcn_wmma_f32_16x16x32_bf16(
                false, a, false, b, (short)0, acc[nt], false, false);
        }
    }
    acc_to_lds(A.b1, true);
    __syncthreads();

    // ---------------- layer 2: [16 x 128] @ [128 x 128] ------------------
    zero_acc();
    for (int kb = 0; kb < 128; kb += 32) {
        bf16x16 a = build_a_lds(kb);
#pragma unroll
        for (int nt = 0; nt < 8; ++nt) {
            bf16x16 b;
#pragma unroll
            for (int s = 0; s < 16; ++s)
                b[s] = A.W2[(size_t)(kb + 8 * h + (s < 8 ? s : 8 + s)) * 128 + nt * 16 + m];
            acc[nt] = __builtin_amdgcn_wmma_f32_16x16x32_bf16(
                false, a, false, b, (short)0, acc[nt], false, false);
        }
    }
    __syncthreads();
    acc_to_lds(A.b2, true);
    __syncthreads();

    // ---------------- layer 3: [16 x 128] @ [128 x 16*NT3] ----------------
    zero_acc();
    for (int kb = 0; kb < 128; kb += 32) {
        bf16x16 a = build_a_lds(kb);
#pragma unroll
        for (int nt = 0; nt < NT3; ++nt) {
            bf16x16 b;
#pragma unroll
            for (int s = 0; s < 16; ++s)
                b[s] = A.W3[(size_t)(kb + 8 * h + (s < 8 ? s : 8 + s)) * (NT3 * 16) + nt * 16 + m];
            acc[nt] = __builtin_amdgcn_wmma_f32_16x16x32_bf16(
                false, a, false, b, (short)0, acc[nt], false, false);
        }
    }

    if constexpr (LN) {
        __syncthreads();
        acc_to_lds(A.b3, false);
        __syncthreads();

        // lane pair (m, m+16) reduces row m; halves combined via shfl_xor(16)
        const int cb = h * 64;
        const float* smr = &sm[m * LSTR + cb];
        float s1 = 0.f, s2 = 0.f;
#pragma unroll
        for (int c4 = 0; c4 < 64; c4 += 4) {
            f32x4 x = *(const f32x4*)(smr + c4);
#pragma unroll
            for (int t = 0; t < 4; ++t) { s1 += x[t]; s2 += x[t] * x[t]; }
        }
        s1 += __shfl_xor(s1, 16);
        s2 += __shfl_xor(s2, 16);
        float mean = s1 * (1.f / 128.f);
        float var  = s2 * (1.f / 128.f) - mean * mean;
        float rs   = rsqrtf(var + 1e-5f);

        int orow = tile0 + m;
        if (orow < A.rows) {
            int sc = 0;
            if constexpr (MODE == 2) sc = A.scat[orow];
            float* po = A.out + (size_t)orow * A.out_ld + cb;
            float* pa = nullptr;
            if constexpr (MODE == 2) pa = A.agg + (size_t)sc * LATENT + cb;
#pragma unroll
            for (int c4 = 0; c4 < 64; c4 += 4) {
                f32x4 x  = *(const f32x4*)(smr + c4);
                f32x4 g  = *(const f32x4*)(A.ln_g + cb + c4);
                f32x4 bb = *(const f32x4*)(A.ln_b + cb + c4);
                f32x4 y;
#pragma unroll
                for (int t = 0; t < 4; ++t) y[t] = (x[t] - mean) * rs * g[t] + bb[t];
                if constexpr (MODE == 0) {
                    *(f32x4*)(po + c4) = y;
                } else {
                    f32x4 old = *(const f32x4*)(po + c4);
#pragma unroll
                    for (int t = 0; t < 4; ++t) old[t] += y[t];
                    *(f32x4*)(po + c4) = old;
                    if constexpr (MODE == 2) {
#pragma unroll
                        for (int t = 0; t < 4; ++t) atomicAdd(pa + c4 + t, y[t]);
                    }
                }
            }
        }
    } else {
        // decoder: small N3, direct store from accumulator 0
#pragma unroll
        for (int v = 0; v < 8; ++v) {
            int row = tile0 + v + 8 * h;
            if (m < A.N3 && row < A.rows)
                A.out[(size_t)row * A.out_ld + m] = acc[0][v] + A.b3[m];
        }
    }
}

// ---------------------------------------------------------------------------
// Host orchestration.
// d_in: 8 data inputs (setup_inputs order), then params flattened as a JAX
// pytree with sorted dict keys: decoder, mesh_enc, node_enc, steps[0..2]
// (edge_mesh, edge_world, node), world_enc. LN blocks: ln_b, ln_g, W1,b1,W2,b2,W3,b3.
// ---------------------------------------------------------------------------
enum {
    IN_FLUID_ATTR = 0, IN_ENV_ATTR, IN_MESH_ATTR, IN_WORLD_ATTR,
    IN_MESH_S, IN_MESH_R, IN_WORLD_S, IN_WORLD_R,
    P_DEC = 8,         // W1 index
    P_MESH_ENC = 14,   // ln_b index
    P_NODE_ENC = 22,
    P_STEPS = 30,      // +24*i per step
    P_WORLD_ENC = 102,
};

enum { V_ENC, V_EDGE, V_NODE, V_DEC };

extern "C" void kernel_launch(void* const* d_in, const int* in_sizes, int n_in,
                              void* d_out, int out_size, void* d_ws, size_t ws_size,
                              hipStream_t stream) {
    (void)in_sizes; (void)n_in; (void)out_size; (void)ws_size;

    const int N_FLUID = 40000, N_ENV = 10000, E_MESH = 200000, E_WORLD = 80000;

    const float* fluid_attr = (const float*)d_in[IN_FLUID_ATTR];
    const float* env_attr   = (const float*)d_in[IN_ENV_ATTR];
    const float* mesh_attr  = (const float*)d_in[IN_MESH_ATTR];
    const float* world_attr = (const float*)d_in[IN_WORLD_ATTR];
    const int* mesh_s  = (const int*)d_in[IN_MESH_S];
    const int* mesh_r  = (const int*)d_in[IN_MESH_R];
    const int* world_s = (const int*)d_in[IN_WORLD_S];
    const int* world_r = (const int*)d_in[IN_WORLD_R];

    // ---- workspace carve-out ----
    char* ws = (char*)d_ws;
    size_t cur = 0;
    auto alloc = [&](size_t bytes) -> char* {
        char* p = ws + cur;
        cur = (cur + bytes + 255) & ~(size_t)255;
        return p;
    };
    float* fluid_lat = (float*)alloc((size_t)N_FLUID * LATENT * 4);
    float* env_lat   = (float*)alloc((size_t)N_ENV   * LATENT * 4);
    float* e_m       = (float*)alloc((size_t)E_MESH  * LATENT * 4);
    float* e_w       = (float*)alloc((size_t)E_WORLD * LATENT * 4);
    float* agg_m     = (float*)alloc((size_t)N_FLUID * LATENT * 4);
    float* agg_w     = (float*)alloc((size_t)N_FLUID * LATENT * 4);

    // ---- block table: 13 MLP blocks ----
    struct BlkDef { int wbase; int K1; int N3; bool ln; };
    BlkDef bd[13];
    bd[0] = { P_NODE_ENC + 2,  12, 128, true };   // node_enc
    bd[1] = { P_MESH_ENC + 2,   7, 128, true };   // mesh_enc
    bd[2] = { P_WORLD_ENC + 2,  4, 128, true };   // world_enc
    bd[3] = { P_DEC,          128,   3, false };  // decoder
    for (int i = 0; i < 3; ++i) {
        bd[4 + 3*i + 0] = { P_STEPS + 24*i + 2,  384, 128, true };  // edge_mesh
        bd[4 + 3*i + 1] = { P_STEPS + 24*i + 10, 384, 128, true };  // edge_world
        bd[4 + 3*i + 2] = { P_STEPS + 24*i + 18, 384, 128, true };  // node
    }

    __bf16* Wb[13][3];
    int Kpad1[13], Npad3[13];
    for (int b = 0; b < 13; ++b) {
        Kpad1[b] = (bd[b].K1 + 31) & ~31;
        Npad3[b] = (bd[b].N3 + 15) & ~15;
        Wb[b][0] = (__bf16*)alloc((size_t)Kpad1[b] * 128 * 2);
        Wb[b][1] = (__bf16*)alloc((size_t)128 * 128 * 2);
        Wb[b][2] = (__bf16*)alloc((size_t)128 * Npad3[b] * 2);
    }

    // ---- pack weights to bf16 ----
    PrepArgs pa;
    int j = 0;
    for (int b = 0; b < 13; ++b) {
        int w = bd[b].wbase;
        pa.jobs[j++] = { (const float*)d_in[w + 0], Wb[b][0], bd[b].K1, 128, Kpad1[b], 128 };
        pa.jobs[j++] = { (const float*)d_in[w + 2], Wb[b][1], 128, 128, 128, 128 };
        pa.jobs[j++] = { (const float*)d_in[w + 4], Wb[b][2], 128, bd[b].N3, 128, Npad3[b] };
    }
    prep_weights_kernel<<<dim3(NJOBS, 32), 256, 0, stream>>>(pa);

    // ---- block launcher: dispatch to specialized instantiation ----
    auto run_block = [&](int variant, int b, int rows, Seg s0, Seg s1, Seg s2,
                         float* out, int out_ld, float* agg, const int* scat) {
        int w = bd[b].wbase;
        BlockArgs A;
        A.seg[0] = s0; A.seg[1] = s1; A.seg[2] = s2;
        A.rows = rows;
        A.W1 = Wb[b][0]; A.W2 = Wb[b][1]; A.W3 = Wb[b][2];
        A.b1 = (const float*)d_in[w + 1];
        A.b2 = (const float*)d_in[w + 3];
        A.b3 = (const float*)d_in[w + 5];
        if (bd[b].ln) { A.ln_b = (const float*)d_in[w - 2]; A.ln_g = (const float*)d_in[w - 1]; }
        else          { A.ln_b = nullptr; A.ln_g = nullptr; }
        A.N3 = bd[b].N3;
        A.out = out; A.out_ld = out_ld; A.agg = agg; A.scat = scat;
        int blocks = (rows + (WAVES_PER_BLOCK * 16) - 1) / (WAVES_PER_BLOCK * 16);
        switch (variant) {
        case V_ENC:  mlp_block_kernel<0,  32, 8, true,  0><<<blocks, TB, 0, stream>>>(A); break;
        case V_EDGE: mlp_block_kernel<1, 384, 8, true,  2><<<blocks, TB, 0, stream>>>(A); break;
        case V_NODE: mlp_block_kernel<1, 384, 8, true,  1><<<blocks, TB, 0, stream>>>(A); break;
        case V_DEC:  mlp_block_kernel<2, 128, 1, false, 0><<<blocks, TB, 0, stream>>>(A); break;
        }
    };

    Seg nil{ nullptr, nullptr, 0, 0, 0 };

    // ---- encoders ----
    run_block(V_ENC, 0, N_FLUID, Seg{ fluid_attr, nullptr, 0, 12, 12 }, nil, nil,
              fluid_lat, LATENT, nullptr, nullptr);
    run_block(V_ENC, 0, N_ENV,   Seg{ env_attr,   nullptr, 0, 12, 12 }, nil, nil,
              env_lat,   LATENT, nullptr, nullptr);
    run_block(V_ENC, 1, E_MESH,  Seg{ mesh_attr,  nullptr, 0,  7,  7 }, nil, nil,
              e_m,       LATENT, nullptr, nullptr);
    run_block(V_ENC, 2, E_WORLD, Seg{ world_attr, nullptr, 0,  4,  4 }, nil, nil,
              e_w,       LATENT, nullptr, nullptr);

    // ---- message-passing steps ----
    for (int i = 0; i < 3; ++i) {
        fill_zero_kernel<<<2048, 256, 0, stream>>>(agg_m, (long)N_FLUID * LATENT);
        fill_zero_kernel<<<2048, 256, 0, stream>>>(agg_w, (long)N_FLUID * LATENT);

        // edge_mesh: [fluid[s], fluid[r], e_m] -> e_m += y ; agg_m[r] += y
        run_block(V_EDGE, 4 + 3*i + 0, E_MESH,
                  Seg{ fluid_lat, mesh_s,  0,   LATENT, LATENT },
                  Seg{ fluid_lat, mesh_r,  128, LATENT, LATENT },
                  Seg{ e_m,       nullptr, 256, LATENT, LATENT },
                  e_m, LATENT, agg_m, mesh_r);

        // edge_world: [env[s], fluid[r], e_w] -> e_w += y ; agg_w[r] += y
        run_block(V_EDGE, 4 + 3*i + 1, E_WORLD,
                  Seg{ env_lat,   world_s, 0,   LATENT, LATENT },
                  Seg{ fluid_lat, world_r, 128, LATENT, LATENT },
                  Seg{ e_w,       nullptr, 256, LATENT, LATENT },
                  e_w, LATENT, agg_w, world_r);

        // node: [fluid, agg_m, agg_w] -> fluid += y
        run_block(V_NODE, 4 + 3*i + 2, N_FLUID,
                  Seg{ fluid_lat, nullptr, 0,   LATENT, LATENT },
                  Seg{ agg_m,     nullptr, 128, LATENT, LATENT },
                  Seg{ agg_w,     nullptr, 256, LATENT, LATENT },
                  fluid_lat, LATENT, nullptr, nullptr);
    }

    // ---- decoder -> d_out (40000 x 3 fp32) ----
    run_block(V_DEC, 3, N_FLUID, Seg{ fluid_lat, nullptr, 0, LATENT, LATENT }, nil, nil,
              (float*)d_out, 3, nullptr, nullptr);
}
